// Actor_43800076484742
// MI455X (gfx1250) — matrix operand order repaired and checked by reading, not compiled, not run on hardware
//
#include <hip/hip_runtime.h>
#include <hip/hip_bf16.h>
#include <math.h>

#define NN 4096
#define DD 256
#define KK 4

typedef __attribute__((ext_vector_type(8)))  float  v8f;
typedef __attribute__((ext_vector_type(8)))  __bf16 v8bf;
typedef __attribute__((ext_vector_type(16))) __bf16 v16bf;

// Split-precision helpers: x ~= hi + lo with ~16 effective mantissa bits.
__device__ inline __bf16 bf_hi(float x) { return (__bf16)x; }
__device__ inline __bf16 bf_lo(float x) { return (__bf16)(x - (float)((__bf16)x)); }

// Concatenate two 8-wide bf16 fragments into one 16-wide WMMA operand.
__device__ inline v16bf cat16(v8bf a, v8bf b) {
  return __builtin_shufflevector(a, b, 0, 1, 2, 3, 4, 5, 6, 7,
                                       8, 9, 10, 11, 12, 13, 14, 15);
}

// ---------------------------------------------------------------------------
// Kernel 0: transpose attributes [N,D] f32 -> attrT_hi/lo [D,N] bf16.
// Tiny (4MB in / 4MB out); makes EA's B-fragments contiguous b128 loads.
// ---------------------------------------------------------------------------
__global__ void conv_attrT(const float* __restrict__ attr,
                           __bf16* __restrict__ attrT_hi,
                           __bf16* __restrict__ attrT_lo) {
  const int k = blockIdx.x * 256 + threadIdx.x;   // 0..N-1 (contiguous writes)
  const int n = blockIdx.y;                       // 0..D-1
  const float a = attr[(size_t)k * DD + n];
  attrT_hi[(size_t)n * NN + k] = bf_hi(a);
  attrT_lo[(size_t)n * NN + k] = bf_lo(a);
}

// ---------------------------------------------------------------------------
// Kernel 1: EA = edges @ attributes via bf16 WMMA 16x16x32.
// edges are exactly 0/1 -> bf16-exact, converted in-register (A fragment).
// attributes use hi+lo split (B fragment): 2 WMMAs per K=32 chunk, routed to
// independent accumulators to avoid the WMMA->WMMA accumulator hazard NOPs.
// ---------------------------------------------------------------------------
__global__ void ea_gemm(const float* __restrict__ edges,
                        const __bf16* __restrict__ attrT_hi,
                        const __bf16* __restrict__ attrT_lo,
                        float* __restrict__ EA) {
  const int lane = threadIdx.x & 31;
  const int wave = threadIdx.x >> 5;
  const int tile = blockIdx.x * 8 + wave;      // (N/16)*(D/16) = 4096 tiles
  const int TN = DD / 16;
  const int tm = tile / TN;
  const int tn = tile % TN;
  const int l15 = lane & 15;

  const float*  ap  = edges + (size_t)(tm * 16 + l15) * NN;    // A row (f32)
  const __bf16* bph = attrT_hi + (size_t)(tn * 16 + l15) * NN; // B col n (bf16)
  const __bf16* bpl = attrT_lo + (size_t)(tn * 16 + l15) * NN;

  const int aoff = (lane < 16) ? 0 : 16;   // A: K split at lane 16 (contiguous k = 16h + s, same as B)
  const int boff = (lane < 16) ? 0 : 16;   // B: K split at lane 16

  v8f acc0 = {}, acc1 = {};
  for (int k0 = 0; k0 < NN; k0 += 32) {
    // A fragment: edges[row][k0+aoff .. +7] and [k0+aoff+8 .. +15], f32->bf16
    v16bf a;
#pragma unroll
    for (int j = 0; j < 8; ++j) a[j] = (__bf16)ap[k0 + aoff + j];
#pragma unroll
    for (int j = 0; j < 8; ++j) a[8 + j] = (__bf16)ap[k0 + aoff + 8 + j];

    // B fragments: 16 contiguous bf16 (32B) per lane
    v16bf bh = *(const v16bf*)(bph + k0 + boff);
    v16bf bl = *(const v16bf*)(bpl + k0 + boff);

    acc0 = __builtin_amdgcn_wmma_f32_16x16x32_bf16(false, a, false, bh,
                                                   (short)0, acc0, false, false);
    acc1 = __builtin_amdgcn_wmma_f32_16x16x32_bf16(false, a, false, bl,
                                                   (short)0, acc1, false, false);
  }

  const int rbase = tm * 16 + ((lane < 16) ? 0 : 8);
  const int coln  = tn * 16 + l15;
#pragma unroll
  for (int v = 0; v < 8; ++v)
    EA[(size_t)(rbase + v) * DD + coln] = acc0[v] + acc1[v];
}

// ---------------------------------------------------------------------------
// Kernel 2: per-row pass. One block per row (256 threads = D columns).
// Emits nf1 as bf16 hi/lo (WMMA-ready) + n2 row norms, plus all [N,D] outputs.
// ---------------------------------------------------------------------------
__global__ void prep(const float* __restrict__ attr,
                     const float* __restrict__ EA,
                     const float* __restrict__ persona,
                     const float* __restrict__ r,
                     const float* __restrict__ W,
                     const int*   __restrict__ d_times,
                     __bf16* __restrict__ nf1_hi,  // [K][N][D]
                     __bf16* __restrict__ nf1_lo,  // [K][N][D]
                     float*  __restrict__ n2,      // [K][N]
                     float* __restrict__ out_attr_prob,
                     float* __restrict__ out_feat_sig,
                     float* __restrict__ out_next_feat,
                     float* __restrict__ out_scaled_attr,
                     float* __restrict__ out_scaled_neigh) {
  const int row = blockIdx.x;
  const int d   = threadIdx.x;
  const int t   = d_times[0];

  const float a  = attr[(size_t)row * DD + d];
  const float ea = EA  [(size_t)row * DD + d];

  __shared__ float red[256];
  float aprob = 0.0f;

  for (int i = 0; i < KK; ++i) {
    const float ri = r[i];
    const float Wi = W[i];
    const float neigh = ea * Wi;
    const float nf = ri * a + (1.0f - ri) * neigh;

    red[d] = nf * nf;
    __syncthreads();
    for (int s = 128; s > 0; s >>= 1) {
      if (d < s) red[d] += red[d + s];
      __syncthreads();
    }
    const float ssum = red[0];
    __syncthreads();

    const float inv = 1.0f / sqrtf(ssum + 1e-10f);
    const float v1  = nf * inv;
    const size_t idx = ((size_t)i * NN + row) * DD + d;
    nf1_hi[idx] = bf_hi(v1);
    nf1_lo[idx] = bf_lo(v1);
    if (d == 0) {
      const float q = ssum / (ssum + 1e-10f);   // ||nf1_row||^2
      n2[i * NN + row] = sqrtf(q + 1e-10f);     // renorm factor for nf2
    }

    const float fs = (nf != 0.0f) ? (1.0f / (1.0f + expf(-nf))) : 0.0f;
    const float pv = persona[((size_t)t * NN + row) * KK + i];
    aprob += fs * pv;

    if (i == KK - 1) {
      out_feat_sig    [(size_t)row * DD + d] = fs;
      out_next_feat   [(size_t)row * DD + d] = nf;
      out_scaled_attr [(size_t)row * DD + d] = ri * a;
      out_scaled_neigh[(size_t)row * DD + d] = (1.0f - ri) * neigh;
    }
  }
  out_attr_prob[(size_t)row * DD + d] = aprob;
}

// ---------------------------------------------------------------------------
// Kernel 3: fused sim GEMM (split-bf16: hi*hi + hi*lo + lo*hi) + masked
// exp/tanh + edges_prob accumulation over all 4 personas in registers.
// The three products per chunk feed three independent accumulators so
// same-accumulator WMMA reuse distance covers the 5-slot bf16 hazard window.
// Block = 256 threads = 8 waves on a 64x32 tile (4x2 sub-tiles of 16x16).
// nf1 hi/lo (16MB) stays L2-resident; sim never touches memory.
// ---------------------------------------------------------------------------
__global__ void sim_fused(const __bf16* __restrict__ nf1_hi,
                          const __bf16* __restrict__ nf1_lo,
                          const float* __restrict__ n2,
                          const float* __restrict__ edges,
                          const float* __restrict__ two_hop,
                          const float* __restrict__ persona,
                          const float* __restrict__ T,
                          const float* __restrict__ e,
                          const int*   __restrict__ d_times,
                          float* __restrict__ edges_prob) {
  const int lane = threadIdx.x & 31;
  const int wave = threadIdx.x >> 5;
  const int rowb = blockIdx.y * 64 + (wave >> 1) * 16;
  const int colb = blockIdx.x * 32 + (wave & 1) * 16;
  const int t    = d_times[0];
  const int l15  = lane & 15;
  const int aoff = (lane < 16) ? 0 : 16;
  const int boff = (lane < 16) ? 0 : 16;
  const int rbase = rowb + ((lane < 16) ? 0 : 8);
  const int c     = colb + l15;

  // i-invariant masks for this lane's 8 output elements
  unsigned adjbits = 0u, discbits = 0u;
#pragma unroll
  for (int v = 0; v < 8; ++v) {
    const float ev = edges  [(size_t)(rbase + v) * NN + c];
    const float th = two_hop[(size_t)(rbase + v) * NN + c];
    const bool adj = (ev != 0.0f);
    adjbits  |= adj ? (1u << v) : 0u;
    discbits |= ((th != 0.0f) && !adj) ? (1u << v) : 0u;
  }

  float eprob[8];
#pragma unroll
  for (int v = 0; v < 8; ++v) eprob[v] = 0.0f;

  for (int i = 0; i < KK; ++i) {
    const size_t base = (size_t)i * NN * DD;
    const __bf16* arh = nf1_hi + base + (size_t)(rowb + l15) * DD;
    const __bf16* arl = nf1_lo + base + (size_t)(rowb + l15) * DD;
    const __bf16* bch = nf1_hi + base + (size_t)(colb + l15) * DD;
    const __bf16* bcl = nf1_lo + base + (size_t)(colb + l15) * DD;

    v8f acc0 = {}, acc1 = {}, acc2 = {};
#pragma unroll
    for (int k0 = 0; k0 < DD; k0 += 32) {
      // A fragments: [k0+aoff .. +7] ++ [k0+aoff+8 .. +15]
      v16bf ah = cat16(*(const v8bf*)(arh + k0 + aoff),
                       *(const v8bf*)(arh + k0 + aoff + 8));
      v16bf al = cat16(*(const v8bf*)(arl + k0 + aoff),
                       *(const v8bf*)(arl + k0 + aoff + 8));
      // B fragments: 16 contiguous bf16
      v16bf bh = *(const v16bf*)(bch + k0 + boff);
      v16bf bl = *(const v16bf*)(bcl + k0 + boff);

      acc0 = __builtin_amdgcn_wmma_f32_16x16x32_bf16(false, ah, false, bh,
                                                     (short)0, acc0, false, false);
      acc1 = __builtin_amdgcn_wmma_f32_16x16x32_bf16(false, ah, false, bl,
                                                     (short)0, acc1, false, false);
      acc2 = __builtin_amdgcn_wmma_f32_16x16x32_bf16(false, al, false, bh,
                                                     (short)0, acc2, false, false);
    }

    const float invT = 1.0f / T[i];
    const float ei   = e[i];
    const float pv   = persona[((size_t)t * NN + c) * KK + i];
    const float n2c  = n2[i * NN + c];

#pragma unroll
    for (int v = 0; v < 8; ++v) {
      const float sim = acc0[v] + acc1[v] + acc2[v];
      const float n2r = n2[i * NN + rbase + v];
      const bool adj  = (adjbits  >> v) & 1u;
      const bool disc = (discbits >> v) & 1u;
      const bool simnz = (sim != 0.0f);

      const float create = (disc && simnz)
                               ? tanhf(expf(sim * invT) * ei) : 0.0f;

      const float sim2   = sim / (n2r * n2c);
      const bool onem    = adj && (sim2 != 0.0f);
      const float dissim = onem ? (1.0f - sim2) : 0.0f;
      const float del    = (onem && simnz)
                               ? tanhf(expf(dissim * sim * invT) * ei) : 0.0f;

      eprob[v] += pv * (create + del);
    }
  }

#pragma unroll
  for (int v = 0; v < 8; ++v)
    edges_prob[(size_t)(rbase + v) * NN + c] = eprob[v];
}

// ---------------------------------------------------------------------------
extern "C" void kernel_launch(void* const* d_in, const int* in_sizes, int n_in,
                              void* d_out, int out_size, void* d_ws, size_t ws_size,
                              hipStream_t stream) {
  const float* attributes = (const float*)d_in[0];
  const float* edges      = (const float*)d_in[1];
  const float* two_hop    = (const float*)d_in[2];
  const float* persona    = (const float*)d_in[3];
  const float* T          = (const float*)d_in[4];
  const float* e          = (const float*)d_in[5];
  const float* r          = (const float*)d_in[6];
  const float* W          = (const float*)d_in[7];
  const int*   times      = (const int*)  d_in[8];

  float* out = (float*)d_out;
  float* out_edges_prob   = out;                                   // [N,N]
  float* out_attr_prob    = out + (size_t)NN * NN;                 // [N,D]
  float* out_feat_sig     = out_attr_prob    + (size_t)NN * DD;    // [N,D]
  float* out_next_feat    = out_feat_sig     + (size_t)NN * DD;    // [N,D]
  float* out_scaled_attr  = out_next_feat    + (size_t)NN * DD;    // [N,D]
  float* out_scaled_neigh = out_scaled_attr  + (size_t)NN * DD;    // [N,D]

  // Workspace carve-up (f32 first, then bf16; all chunk sizes 128B-multiples)
  float*  ws_EA  = (float*)d_ws;                                   // [N,D] f32
  float*  ws_n2  = ws_EA + (size_t)NN * DD;                        // [K,N] f32
  __bf16* ws_aTh = (__bf16*)(ws_n2 + (size_t)KK * NN);             // [D,N]
  __bf16* ws_aTl = ws_aTh + (size_t)DD * NN;                       // [D,N]
  __bf16* ws_nh  = ws_aTl + (size_t)DD * NN;                       // [K,N,D]
  __bf16* ws_nl  = ws_nh  + (size_t)KK * NN * DD;                  // [K,N,D]

  // 0) attributes -> transposed bf16 hi/lo
  conv_attrT<<<dim3(NN / 256, DD), 256, 0, stream>>>(attributes, ws_aTh, ws_aTl);

  // 1) EA = edges @ attributes (bf16 WMMA, edges exact in bf16)
  ea_gemm<<<512, 256, 0, stream>>>(edges, ws_aTh, ws_aTl, ws_EA);

  // 2) per-row: nf1 hi/lo + n2 + all [N,D] outputs
  prep<<<NN, 256, 0, stream>>>(attributes, ws_EA, persona, r, W, times,
                               ws_nh, ws_nl, ws_n2,
                               out_attr_prob, out_feat_sig, out_next_feat,
                               out_scaled_attr, out_scaled_neigh);

  // 3) fused split-bf16 sim GEMM + edges_prob
  dim3 grid(NN / 32, NN / 64);
  sim_fused<<<grid, 256, 0, stream>>>(ws_nh, ws_nl, ws_n2, edges, two_hop,
                                      persona, T, e, times, out_edges_prob);

  (void)in_sizes; (void)n_in; (void)out_size; (void)ws_size;
}
